// My_SA_88098369175821
// MI455X (gfx1250) — matrix operand order repaired and checked by reading, not compiled, not run on hardware
//
#include <hip/hip_runtime.h>
#include <math.h>

// Problem constants
#define BATCH 4096
#define TT 128         // sequence length
#define CC 20          // channels
#define KQ 100         // key/query dim
#define SCALE 0.08838834764831845f   // 1/sqrt(128)

typedef _Float16 v16h __attribute__((ext_vector_type(16)));
typedef _Float16 v8h  __attribute__((ext_vector_type(8)));
typedef float    v8f  __attribute__((ext_vector_type(8)));

// ---- LDS layout (byte offsets into dynamic shared) -------------------------
// xs  : f32 [128][20]            10240
// xh  : f16 [128][32]  (K pad)    8192
// WkT : f16 [112][32]             7168
// WqT : f16 [112][32]             7168
// WvT : f16 [32][32]              2048
// vT  : f16 [32][128]             8192   (v transposed: vT[d][j] = v[j][d])
// kS  : f16 [128][128] (K pad)   32768
// qS  : f16 [128][128] (K pad)   32768   (reused as softmaxed w after scores)
// sS  : f32 [128][130] (padded)  66560
// bkS : f32 [112] (zero pad)       448
// bqS : f32 [112]                  448
// bvS : f32 [32]                   128
#define OFF_XS   0
#define OFF_XH   10240
#define OFF_WKT  18432
#define OFF_WQT  25600
#define OFF_WVT  32768
#define OFF_VT   34816
#define OFF_KS   43008
#define OFF_QS   75776
#define OFF_SS   108544
#define OFF_BK   175104
#define OFF_BQ   175552
#define OFF_BV   176000
#define SMEM_BYTES 176128
// region to pre-zero (covers xh..qS, all the f16 padded buffers)
#define ZERO_OFF 10240
#define ZERO_LEN 98304

// A fragment (16x32 f16, M x K): lanes 0-15 row M=lane, halves 0..7 -> K=k0+0..7,
// halves 8..15 -> K=k0+16..23; lanes 16-31 row M=lane-16, K ranges +8.
__device__ __forceinline__ v16h load_frag_a(const _Float16* buf, int stride,
                                            int m0, int k0, int lane) {
    int m    = m0 + (lane & 15);
    int koff = (lane & 16) ? 8 : 0;
    const _Float16* p = buf + m * stride + k0 + koff;
    v8h lo = *reinterpret_cast<const v8h*>(p);
    v8h hi = *reinterpret_cast<const v8h*>(p + 16);
    v16h a;
#pragma unroll
    for (int i = 0; i < 8; ++i) { a[i] = lo[i]; a[i + 8] = hi[i]; }
    return a;
}

// B fragment (32x16 f16, K x N) read from an [N][K]-major f16 buffer:
// lane 0-15: N=n0+lane, K=k0+0..7 then k0+16..23; lanes 16-31: K ranges +8.
__device__ __forceinline__ v16h load_frag_b(const _Float16* buf, int stride,
                                            int n0, int k0, int lane) {
    int n  = n0 + (lane & 15);
    int kb = k0 + ((lane & 16) ? 8 : 0);
    const _Float16* p = buf + n * stride + kb;
    v8h lo = *reinterpret_cast<const v8h*>(p);
    v8h hi = *reinterpret_cast<const v8h*>(p + 16);
    v16h b;
#pragma unroll
    for (int i = 0; i < 8; ++i) { b[i] = lo[i]; b[i + 8] = hi[i]; }
    return b;
}

__device__ __forceinline__ v8f wmma32(v16h a, v16h b, v8f c) {
    return __builtin_amdgcn_wmma_f32_16x16x32_f16(false, a, false, b,
                                                  (short)0, c, false, false);
}

// Branchless activations (no exec-mask diamonds, no IEEE div expansion):
__device__ __forceinline__ float elu_f(float x) {
    float e = __expf(x) - 1.0f;          // exp fast path; discarded when x>0
    return x > 0.0f ? x : e;
}
__device__ __forceinline__ float tanh_f(float x) {
    float xc = fminf(fmaxf(x, -15.0f), 15.0f);   // avoid inf/inf
    float t  = __expf(2.0f * xc);
    return (t - 1.0f) * __builtin_amdgcn_rcpf(t + 1.0f);
}

__global__ __launch_bounds__(256, 1)
void fused_attn_kernel(const float* __restrict__ x,
                       const float* __restrict__ Wk, const float* __restrict__ bk,
                       const float* __restrict__ Wq, const float* __restrict__ bq,
                       const float* __restrict__ Wv, const float* __restrict__ bv,
                       float* __restrict__ out) {
    extern __shared__ char smem[];
    float*    xs  = reinterpret_cast<float*>(smem + OFF_XS);
    _Float16* xh  = reinterpret_cast<_Float16*>(smem + OFF_XH);
    _Float16* WkT = reinterpret_cast<_Float16*>(smem + OFF_WKT);
    _Float16* WqT = reinterpret_cast<_Float16*>(smem + OFF_WQT);
    _Float16* WvT = reinterpret_cast<_Float16*>(smem + OFF_WVT);
    _Float16* vT  = reinterpret_cast<_Float16*>(smem + OFF_VT);
    _Float16* kS  = reinterpret_cast<_Float16*>(smem + OFF_KS);
    _Float16* qS  = reinterpret_cast<_Float16*>(smem + OFF_QS);
    float*    sS  = reinterpret_cast<float*>(smem + OFF_SS);
    float*    bkS = reinterpret_cast<float*>(smem + OFF_BK);
    float*    bqS = reinterpret_cast<float*>(smem + OFF_BQ);
    float*    bvS = reinterpret_cast<float*>(smem + OFF_BV);

    const int tid  = threadIdx.x;
    const int lane = tid & 31;
    const int wid  = tid >> 5;
    const int b    = blockIdx.x;

    // ---- Phase 0: zero the padded f16 staging buffers ----------------------
    {
        int4* z = reinterpret_cast<int4*>(smem + ZERO_OFF);
        const int nvec = ZERO_LEN / 16;
#pragma unroll 4
        for (int i = tid; i < nvec; i += 256) z[i] = make_int4(0, 0, 0, 0);
    }
    __syncthreads();

    // ---- Phase 1: stage x, transposed f16 weights, zero-padded biases ------
    {
        const float* xb = x + (size_t)b * (TT * CC);
        for (int i = tid; i < TT * CC; i += 256) {
            float v = xb[i];
            xs[i] = v;
            int t = i / CC, c = i % CC;
            xh[t * 32 + c] = (_Float16)v;
        }
        for (int i = tid; i < CC * KQ; i += 256) {
            int c = i / KQ, n = i % KQ;
            WkT[n * 32 + c] = (_Float16)Wk[i];
            WqT[n * 32 + c] = (_Float16)Wq[i];
        }
        for (int i = tid; i < CC * CC; i += 256) {
            int c = i / CC, n = i % CC;
            WvT[n * 32 + c] = (_Float16)Wv[i];
        }
        if (tid < 112) {
            bkS[tid] = (tid < KQ) ? bk[tid] : 0.0f;
            bqS[tid] = (tid < KQ) ? bq[tid] : 0.0f;
        }
        if (tid < 32) bvS[tid] = (tid < CC) ? bv[tid] : 0.0f;
    }
    __syncthreads();

    // ---- Phase 2: projections (statically specialized loops) ---------------
    // k = elu(x@Wk + bk): 56 tiles (8x7), 7 per wave
    for (int t = wid; t < 56; t += 8) {
        int m0 = (t / 7) * 16, n0 = (t % 7) * 16;
        v16h a = load_frag_a(xh, 32, m0, 0, lane);
        v16h bb = load_frag_b(WkT, 32, n0, 0, lane);
        v8f acc = {};
        acc = wmma32(a, bb, acc);
        int n = n0 + (lane & 15);
        float bias_v = bkS[n];
        int mofs = (lane & 16) ? 8 : 0;
#pragma unroll
        for (int r = 0; r < 8; ++r) {
            float val = elu_f(acc[r] + bias_v);
            if (n < KQ) kS[(m0 + r + mofs) * 128 + n] = (_Float16)val;
        }
    }
    // q = elu(x@Wq + bq): 56 tiles, 7 per wave
    for (int t = wid; t < 56; t += 8) {
        int m0 = (t / 7) * 16, n0 = (t % 7) * 16;
        v16h a = load_frag_a(xh, 32, m0, 0, lane);
        v16h bb = load_frag_b(WqT, 32, n0, 0, lane);
        v8f acc = {};
        acc = wmma32(a, bb, acc);
        int n = n0 + (lane & 15);
        float bias_v = bqS[n];
        int mofs = (lane & 16) ? 8 : 0;
#pragma unroll
        for (int r = 0; r < 8; ++r) {
            float val = elu_f(acc[r] + bias_v);
            if (n < KQ) qS[(m0 + r + mofs) * 128 + n] = (_Float16)val;
        }
    }
    // v = tanh(x@Wv + bv): 16 tiles (8x2), 2 per wave, stored transposed
    for (int t = wid; t < 16; t += 8) {
        int m0 = (t / 2) * 16, n0 = (t % 2) * 16;
        v16h a = load_frag_a(xh, 32, m0, 0, lane);
        v16h bb = load_frag_b(WvT, 32, n0, 0, lane);
        v8f acc = {};
        acc = wmma32(a, bb, acc);
        int n = n0 + (lane & 15);
        float bias_v = bvS[n];
        int mofs = (lane & 16) ? 8 : 0;
#pragma unroll
        for (int r = 0; r < 8; ++r) {
            float val = tanh_f(acc[r] + bias_v);
            if (n < CC) vT[n * 128 + (m0 + r + mofs)] = (_Float16)val;  // vT[d][j]
        }
    }
    __syncthreads();

    // ---- Phase 3: scores s[i][j] = SCALE * sum_d k[i][d]*q[j][d] -----------
    // 64 tiles (8x8), 8 per wave, K=128 (100 real + zero pad) in 4 steps
    for (int t = wid; t < 64; t += 8) {
        int m0 = (t >> 3) * 16, n0 = (t & 7) * 16;
        v8f acc = {};
#pragma unroll
        for (int kk = 0; kk < 4; ++kk) {
            int k0 = kk * 32;
            v16h a = load_frag_a(kS, 128, m0, k0, lane);
            v16h bb = load_frag_b(qS, 128, n0, k0, lane);  // qS[j][d] is [N][K]
            acc = wmma32(a, bb, acc);
        }
        int n = n0 + (lane & 15);
        int mofs = (lane & 16) ? 8 : 0;
#pragma unroll
        for (int r = 0; r < 8; ++r)
            sS[(m0 + r + mofs) * 130 + n] = acc[r] * SCALE;
    }
    __syncthreads();

    // ---- Phase 4: softmax over axis=1 (over rows i, per column j) ----------
    if (tid < TT) {
        int j = tid;
        float mx = -3.0e38f;
        for (int i = 0; i < TT; ++i) mx = fmaxf(mx, sS[i * 130 + j]);
        float sum = 0.0f;
        for (int i = 0; i < TT; ++i) {
            float e = __expf(sS[i * 130 + j] - mx);
            sum += e;
            sS[i * 130 + j] = e;
        }
        float inv = 1.0f / sum;
        for (int i = 0; i < TT; ++i)                 // overwrite qS with w (f16)
            qS[i * 128 + j] = (_Float16)(sS[i * 130 + j] * inv);
    }
    __syncthreads();

    // ---- Phase 5: y = w @ v + x  (16 tiles: 8x2, 2 per wave, K=128) --------
    for (int t = wid * 2; t < wid * 2 + 2; ++t) {
        int m0 = (t >> 1) * 16, n0 = (t & 1) * 16;
        v8f acc = {};
#pragma unroll
        for (int kk = 0; kk < 4; ++kk) {
            int k0 = kk * 32;
            v16h a = load_frag_a(qS, 128, m0, k0, lane);   // w[i][j]
            v16h bb = load_frag_b(vT, 128, n0, k0, lane);  // vT[d][j] is [N][K]
            acc = wmma32(a, bb, acc);
        }
        int n = n0 + (lane & 15);
        int mofs = (lane & 16) ? 8 : 0;
        if (n < CC) {
            float* ob = out + (size_t)b * (TT * CC);
#pragma unroll
            for (int r = 0; r < 8; ++r) {
                int m = m0 + r + mofs;
                ob[m * CC + n] = acc[r] + xs[m * CC + n];
            }
        }
    }
}

extern "C" void kernel_launch(void* const* d_in, const int* in_sizes, int n_in,
                              void* d_out, int out_size, void* d_ws, size_t ws_size,
                              hipStream_t stream) {
    (void)in_sizes; (void)n_in; (void)out_size; (void)d_ws; (void)ws_size;
    const float* x  = (const float*)d_in[0];
    const float* Wk = (const float*)d_in[1];
    const float* bk = (const float*)d_in[2];
    const float* Wq = (const float*)d_in[3];
    const float* bq = (const float*)d_in[4];
    const float* Wv = (const float*)d_in[5];
    const float* bv = (const float*)d_in[6];
    float* out = (float*)d_out;

    static bool attr_set = false;
    if (!attr_set) {
        hipFuncSetAttribute((const void*)fused_attn_kernel,
                            hipFuncAttributeMaxDynamicSharedMemorySize, SMEM_BYTES);
        attr_set = true;
    }
    fused_attn_kernel<<<BATCH, 256, SMEM_BYTES, stream>>>(x, Wk, bk, Wq, bq, Wv, bv, out);
}
